// Gating_8658654068957
// MI455X (gfx1250) — matrix operand order repaired and checked by reading, not compiled, not run on hardware
//
#include <hip/hip_runtime.h>
#include <hip/hip_bf16.h>

// ---------------------------------------------------------------------------
// MoE gating for MI455X (gfx1250, wave32, WMMA)
//   logits = x[16384,4096] @ W[64,4096]^T   (bf16 WMMA, f32 accum)
//   per-row top-8 of 64, sparse softmax
// Outputs concat: sparse_probs[16384*64] | indices[16384*8] | gate_logit[16384*64]
//
// Roofline: x stream = 256 MB @ 23.3 TB/s ~= 11.4 us floor; bf16 WMMA compute
// ~3.4 us << floor, so the K-loop is software-pipelined (prefetch next A/B
// fragments into registers before the 4 back-to-back WMMAs) to keep the vmem
// pipe saturated instead of stalling on s_wait_loadcnt 0 per WMMA.
// ---------------------------------------------------------------------------

typedef __bf16 bf16_t;
typedef __attribute__((ext_vector_type(16))) __bf16 v16bf;
typedef __attribute__((ext_vector_type(8)))  float  v8f;
typedef __attribute__((ext_vector_type(4)))  float  v4f;

#define HDIM   4096
#define EXPERTS 64
#define TOPK    8
#define ROWS   16384            // B*S = 4*4096
#define ROWS_PER_BLOCK 128      // 8 waves * 16 rows
#define LDS_PITCH 68            // 64 + pad: rows stay 16B-aligned, no bank conflicts

static constexpr long long SP_OFF  = 0;                          // sparse_probs
static constexpr long long IDX_OFF = (long long)ROWS * EXPERTS;  // indices (as float)
static constexpr long long GL_OFF  = IDX_OFF + (long long)ROWS * TOPK; // gate_logit

// ---------------------------------------------------------------------------
// Kernel 0: one-time W f32 -> bf16 conversion into workspace (512 KB)
// ---------------------------------------------------------------------------
__global__ void gating_wcvt_kernel(const float* __restrict__ W,
                                   bf16_t* __restrict__ Wb, int n) {
    int i = blockIdx.x * blockDim.x + threadIdx.x;
    if (i < n) Wb[i] = (bf16_t)W[i];
}

// ---------------------------------------------------------------------------
// Kernel 1: fused GEMM (bf16 WMMA, software-pipelined) + top-k + sparse softmax
// ---------------------------------------------------------------------------
__launch_bounds__(256)
__global__ void gating_main_kernel(const float* __restrict__ x,
                                   const bf16_t* __restrict__ Wb,
                                   float* __restrict__ out) {
    __shared__ float lds[ROWS_PER_BLOCK][LDS_PITCH];

    const int tid    = threadIdx.x;
    const int wave   = tid >> 5;        // 0..7
    const int lane   = tid & 31;
    const int laneHi = lane >> 4;       // 0 | 1
    const int lane16 = lane & 15;

    const int m0   = blockIdx.x * ROWS_PER_BLOCK + wave * 16;  // first row of this wave
    const int rowA = m0 + lane16;                              // A-fragment row, this lane

    // A: lanes 0-15 hold K = [0..15], lanes 16-31 hold K = [16..31]
    const float*  __restrict__ ap    = x  + (long long)rowA * HDIM + laneHi * 16;
    // B: lane l<16 -> W[e0+l][k0..k0+15], lane l>=16 -> W[e0+l-16][k0+16..k0+31]
    const bf16_t* __restrict__ wbase = Wb + (long long)lane16 * HDIM + laneHi * 16;

    auto loadA = [&](int k0) -> v16bf {
        v4f a0 = *(const v4f*)(ap + k0);
        v4f a1 = *(const v4f*)(ap + k0 + 4);
        v4f a2 = *(const v4f*)(ap + k0 + 8);
        v4f a3 = *(const v4f*)(ap + k0 + 12);
        v16bf a;
        #pragma unroll
        for (int i = 0; i < 4; ++i) {
            a[i]      = (bf16_t)a0[i];
            a[4 + i]  = (bf16_t)a1[i];
            a[8 + i]  = (bf16_t)a2[i];
            a[12 + i] = (bf16_t)a3[i];
        }
        return a;
    };
    auto loadB = [&](int t, int k0) -> v16bf {
        return *(const v16bf*)(wbase + (long long)t * 16 * HDIM + k0);
    };

    v8f acc[4] = {};   // experts [0:16) [16:32) [32:48) [48:64), 16 rows each

    // ---- software-pipelined K loop: prefetch k0+32 while WMMAs run on k0 ----
    v16bf aCur = loadA(0);
    v16bf bCur[4];
    #pragma unroll
    for (int t = 0; t < 4; ++t) bCur[t] = loadB(t, 0);

    #pragma unroll 2
    for (int k0 = 0; k0 < HDIM; k0 += 32) {
        // clamp prefetch on last iteration (harmless redundant load, no branch)
        const int kn = (k0 + 32 < HDIM) ? (k0 + 32) : k0;
        v16bf aNxt = loadA(kn);
        v16bf bNxt[4];
        #pragma unroll
        for (int t = 0; t < 4; ++t) bNxt[t] = loadB(t, kn);

        #pragma unroll
        for (int t = 0; t < 4; ++t) {
            acc[t] = __builtin_amdgcn_wmma_f32_16x16x32_bf16(
                         false, aCur, false, bCur[t], (short)0, acc[t], false, false);
        }

        aCur = aNxt;
        #pragma unroll
        for (int t = 0; t < 4; ++t) bCur[t] = bNxt[t];
    }

    // --- spill logits to LDS (C layout: VGPR r = row r / r+8; lane%16 = col) ---
    const int lrowBase = wave * 16 + laneHi * 8;
    #pragma unroll
    for (int t = 0; t < 4; ++t) {
        const int col = t * 16 + lane16;
        #pragma unroll
        for (int r = 0; r < 8; ++r) {
            lds[lrowBase + r][col] = acc[t][r];
        }
    }
    __syncthreads();

    // --- per-row top-8 + sparse softmax (threads 0..127, one row each) ---
    if (tid < ROWS_PER_BLOCK) {
        const long long grow = (long long)blockIdx.x * ROWS_PER_BLOCK + tid;
        float* lrow = &lds[tid][0];

        // gate_logit: vectorized row copy (row base is 16B-aligned: pitch 68)
        {
            float* gl = out + GL_OFF + grow * EXPERTS;
            #pragma unroll
            for (int j = 0; j < EXPERTS; j += 4) {
                *(v4f*)(gl + j) = *(const v4f*)(lrow + j);
            }
        }

        // top-8 selection, first-index-wins on ties (matches jax.lax.top_k)
        int   tidx[TOPK];
        float tval[TOPK];
        #pragma unroll
        for (int kk = 0; kk < TOPK; ++kk) {
            float best = -INFINITY;
            int   bi   = 0;
            for (int j = 0; j < EXPERTS; ++j) {
                float v = lrow[j];
                if (v > best) { best = v; bi = j; }
            }
            tidx[kk] = bi;
            tval[kk] = best;
            lrow[bi] = -INFINITY;
        }

        // softmax over the 8 kept logits (others are exactly 0 in sparse_probs)
        float mx = tval[0];
        float ex[TOPK], s = 0.f;
        #pragma unroll
        for (int kk = 0; kk < TOPK; ++kk) { ex[kk] = __expf(tval[kk] - mx); s += ex[kk]; }
        const float inv = 1.0f / s;

        // sparse_probs: zero-fill then scatter the 8 probs (same-thread store order kept)
        float* sp = out + SP_OFF + grow * EXPERTS;
        const v4f z = {0.f, 0.f, 0.f, 0.f};
        #pragma unroll
        for (int j = 0; j < EXPERTS; j += 4) *(v4f*)(sp + j) = z;
        #pragma unroll
        for (int kk = 0; kk < TOPK; ++kk) sp[tidx[kk]] = ex[kk] * inv;

        // indices (stored as float: d_out is a single f32 buffer for the tuple)
        float* ip = out + IDX_OFF + grow * TOPK;
        #pragma unroll
        for (int kk = 0; kk < TOPK; ++kk) ip[kk] = (float)tidx[kk];
    }
}

// ---------------------------------------------------------------------------
extern "C" void kernel_launch(void* const* d_in, const int* in_sizes, int n_in,
                              void* d_out, int out_size, void* d_ws, size_t ws_size,
                              hipStream_t stream) {
    const float* x = (const float*)d_in[0];   // [4,4096,4096] f32
    const float* W = (const float*)d_in[1];   // [64,4096] f32
    float* out = (float*)d_out;
    bf16_t* Wb = (bf16_t*)d_ws;               // 64*4096*2 = 512 KB scratch

    const int nW = in_sizes[1];               // 262144
    gating_wcvt_kernel<<<(nW + 255) / 256, 256, 0, stream>>>(W, Wb, nW);

    gating_main_kernel<<<ROWS / ROWS_PER_BLOCK, 256, 0, stream>>>(x, Wb, out);
    (void)n_in; (void)out_size; (void)ws_size;
}
